// VerifyLlamaAttention_75600014344451
// MI455X (gfx1250) — matrix operand order repaired and checked by reading, not compiled, not run on hardware
//
#include <hip/hip_runtime.h>
#include <hip/hip_bf16.h>

// ---------------------------------------------------------------------------
// Llama attention block on gfx1250 (MI455X), bf16 WMMA everywhere.
//   HIDDEN=4096, S=2048, N_HEADS=32, N_KV=8, HEAD_DIM=128
// ---------------------------------------------------------------------------

typedef __attribute__((ext_vector_type(16))) __bf16   v16bf;
typedef __attribute__((ext_vector_type(8)))  __bf16   v8bf;
typedef __attribute__((ext_vector_type(8)))  float    v8f;
typedef __attribute__((ext_vector_type(4)))  unsigned v4u;
typedef __attribute__((ext_vector_type(8)))  int      v8i;
typedef __attribute__((ext_vector_type(4)))  int      v4i;

#define SEQ     2048
#define HID     4096
#define NHEADS  32
#define NKV     8
#define HDIM    128
#define KVDIM   (NKV * HDIM)   // 1024

#if defined(__gfx1250__) && __has_builtin(__builtin_amdgcn_tensor_load_to_lds) && \
    __has_builtin(__builtin_amdgcn_s_wait_tensorcnt)
#define USE_TDM 1
#else
#define USE_TDM 0
#endif

// ---------------------------------------------------------------------------
// WMMA fragment loaders (contiguous k order, same for A and B; not the ISA table order, wave32)
// A (16x32, MxK): lanes 0-15 -> M=lane, elems K0..15;
//                 lanes 16-31 -> M=lane-16, elems K16..31
// B (32x16, KxN): lanes 0-15 -> N=lane, elems K0..15;
//                 lanes 16-31 -> N=lane-16, elems K16..31
// ---------------------------------------------------------------------------

__device__ __forceinline__ v16bf load_frag_a(const __bf16* base, int ld,
                                             int row0, int k0) {
  const int lane = threadIdx.x & 31;
  const int m  = lane & 15;
  const int kA = k0 + ((lane < 16) ? 0 : 16);
  const int kB = kA + 8;
  const __bf16* r = base + (size_t)(row0 + m) * ld;
  union { v16bf v; v8bf h[2]; } u;
  u.h[0] = *reinterpret_cast<const v8bf*>(r + kA);
  u.h[1] = *reinterpret_cast<const v8bf*>(r + kB);
  return u.v;
}

// B[k][n] = W[n][k] (NT GEMM: W row-major with contiguous K)
__device__ __forceinline__ v16bf load_frag_b_nt(const __bf16* base, int ld,
                                                int n0, int k0) {
  const int lane = threadIdx.x & 31;
  const int n  = n0 + (lane & 15);
  const int kb = k0 + ((lane < 16) ? 0 : 16);
  const __bf16* r = base + (size_t)n * ld + kb;
  union { v16bf v; v8bf h[2]; } u;
  u.h[0] = *reinterpret_cast<const v8bf*>(r);
  u.h[1] = *reinterpret_cast<const v8bf*>(r + 8);
  return u.v;
}

// B[k][n] = V[k][n] (V row-major over n: column-gather, used from LDS)
__device__ __forceinline__ v16bf load_frag_b_colmajor(const __bf16* base, int ld,
                                                      int n0, int k0) {
  const int lane = threadIdx.x & 31;
  const int n  = n0 + (lane & 15);
  const int kb = k0 + ((lane < 16) ? 0 : 16);
  v16bf v;
#pragma unroll
  for (int e = 0; e < 16; ++e) v[e] = base[(size_t)(kb + e) * ld + n];
  return v;
}

// ---------------------------------------------------------------------------
// Tensor Data Mover: 2D tile load (global -> LDS), D# per ISA 08 §8.
//   data_size = 2 bytes (bf16); tile rows x cols; row stride in elements.
//   6-arg builtin form (clang-23 / therock-10.0 lane).
// ---------------------------------------------------------------------------
#if USE_TDM
__device__ __forceinline__ void tdm_load_tile_2d(__bf16* lds_dst, const __bf16* gsrc,
                                                 unsigned rows, unsigned cols,
                                                 unsigned stride_elems) {
  const unsigned long long ga = (unsigned long long)(uintptr_t)gsrc;
  const unsigned lds_off = (unsigned)(uintptr_t)lds_dst;   // low 32 bits = LDS offset

  v4u g0;
  g0[0] = 1u;                                     // count=1, user descriptor
  g0[1] = lds_off;                                // lds_addr [63:32]
  g0[2] = (unsigned)(ga & 0xFFFFFFFFu);           // global_addr [95:64]
  g0[3] = (unsigned)((ga >> 32) & 0x1FFFFFFu)     // global_addr [120:96]
        | (2u << 30);                             // type=2 ("image") [127:126]

  v8i g1;
  g1[0] = (int)(1u << 16);                        // wg_mask=0, data_size=1 (2B)
  g1[1] = (int)((cols & 0xFFFFu) << 16);          // tensor_dim0[15:0]
  g1[2] = (int)(((cols >> 16) & 0xFFFFu) | ((rows & 0xFFFFu) << 16)); // dim0 hi | dim1 lo
  g1[3] = (int)(((rows >> 16) & 0xFFFFu) | ((cols & 0xFFFFu) << 16)); // dim1 hi | tile_dim0
  g1[4] = (int)(rows & 0xFFFFu);                  // tile_dim1 | tile_dim2=0
  g1[5] = (int)stride_elems;                      // tensor_dim0_stride[31:0]
  g1[6] = 0;                                      // stride hi | dim1_stride lo
  g1[7] = 0;

  v4i gz4 = {};
  v8i gz8 = {};
  __builtin_amdgcn_tensor_load_to_lds(g0, g1, gz4, gz4, gz8, 0);
}
#endif

// ---------------------------------------------------------------------------
// fp32 -> bf16 elementwise conversion
// ---------------------------------------------------------------------------
__global__ void cvt_f32_bf16(const float* __restrict__ in,
                             __bf16* __restrict__ out, size_t n) {
  size_t i = (size_t)blockIdx.x * blockDim.x + threadIdx.x;
  size_t stride = (size_t)gridDim.x * blockDim.x;
  for (; i < n; i += stride) out[i] = (__bf16)in[i];
}

// ---------------------------------------------------------------------------
// NT GEMM: C[M,N] = A[M,K] * B[N,K]^T ; A,B bf16, accum f32.
// Block = 128 threads = 4 waves in 2x2; each wave owns a 32x64 tile
// (2x4 accumulators => 8 WMMAs per 12 b128 loads per k-step of 32).
// Output either fp32 (Cf) or bf16 (Cb).
// ---------------------------------------------------------------------------
__global__ void __launch_bounds__(128)
gemm_nt_wmma(const __bf16* __restrict__ A, const __bf16* __restrict__ B,
             float* __restrict__ Cf, __bf16* __restrict__ Cb,
             int M, int N, int K) {
  const int lane = threadIdx.x & 31;
  const int wave = threadIdx.x >> 5;
  const int row0 = blockIdx.y * 64  + (wave >> 1) * 32;
  const int col0 = blockIdx.x * 128 + (wave & 1) * 64;

  v8f c[2][4];
#pragma unroll
  for (int i = 0; i < 2; ++i)
#pragma unroll
    for (int j = 0; j < 4; ++j) c[i][j] = (v8f){};

  for (int k = 0; k < K; k += 32) {
    if (k + 256 < K) {   // -> global_prefetch_b8
      __builtin_prefetch(A + (size_t)(row0 + (lane & 15)) * K + k + 256, 0, 1);
      __builtin_prefetch(B + (size_t)(col0 + (lane & 15)) * K + k + 256, 0, 1);
      __builtin_prefetch(B + (size_t)(col0 + 32 + (lane & 15)) * K + k + 256, 0, 1);
    }
    v16bf a0 = load_frag_a(A, K, row0,      k);
    v16bf a1 = load_frag_a(A, K, row0 + 16, k);
    v16bf b[4];
#pragma unroll
    for (int j = 0; j < 4; ++j) b[j] = load_frag_b_nt(B, K, col0 + j * 16, k);
#pragma unroll
    for (int j = 0; j < 4; ++j) {
      c[0][j] = __builtin_amdgcn_wmma_f32_16x16x32_bf16(false, a0, false, b[j], (short)0, c[0][j], false, false);
      c[1][j] = __builtin_amdgcn_wmma_f32_16x16x32_bf16(false, a1, false, b[j], (short)0, c[1][j], false, false);
    }
  }

  // C/D layout: VGPR r -> row r (lanes 0-15) / row r+8 (lanes 16-31), col = lane%16
  const int rb = (lane < 16) ? 0 : 8;
  const int cc = lane & 15;
#pragma unroll
  for (int i = 0; i < 2; ++i) {
#pragma unroll
    for (int r = 0; r < 8; ++r) {
      const size_t row = (size_t)(row0 + i * 16 + rb + r);
#pragma unroll
      for (int j = 0; j < 4; ++j) {
        const size_t idx = row * N + col0 + j * 16 + cc;
        if (Cf) Cf[idx] = c[i][j][r];
        else    Cb[idx] = (__bf16)c[i][j][r];
      }
    }
  }
}

// ---------------------------------------------------------------------------
// Rotary embedding on fp32 Q/K, repacked to bf16.
// One thread per (s, head, d<64); heads 0..31 = Q, heads 32..39 = K.
// ---------------------------------------------------------------------------
__global__ void rope_pack(const float* __restrict__ Qf, const float* __restrict__ Kf,
                          const float* __restrict__ cosp, const float* __restrict__ sinp,
                          __bf16* __restrict__ Qb, __bf16* __restrict__ Kb) {
  int idx = blockIdx.x * blockDim.x + threadIdx.x;
  const int total = SEQ * (NHEADS + NKV) * (HDIM / 2);
  if (idx >= total) return;
  const int d = idx & 63; idx >>= 6;
  const int h = idx % (NHEADS + NKV);
  const int s = idx / (NHEADS + NKV);
  const float c1 = cosp[s * HDIM + d],      s1 = sinp[s * HDIM + d];
  const float c2 = cosp[s * HDIM + d + 64], s2 = sinp[s * HDIM + d + 64];
  if (h < NHEADS) {
    const float* q = Qf + (size_t)s * HID + h * HDIM;
    __bf16*      o = Qb + (size_t)s * HID + h * HDIM;
    const float x1 = q[d], x2 = q[d + 64];
    o[d]      = (__bf16)(x1 * c1 - x2 * s1);
    o[d + 64] = (__bf16)(x2 * c2 + x1 * s2);
  } else {
    const int hh = h - NHEADS;
    const float* kk = Kf + (size_t)s * KVDIM + hh * HDIM;
    __bf16*      o  = Kb + (size_t)s * KVDIM + hh * HDIM;
    const float x1 = kk[d], x2 = kk[d + 64];
    o[d]      = (__bf16)(x1 * c1 - x2 * s1);
    o[d + 64] = (__bf16)(x2 * c2 + x1 * s2);
  }
}

// ---------------------------------------------------------------------------
// Flash attention: grid = (32 heads, 32 q-blocks of 64 rows), 128 threads.
// Each wave owns 16 query rows. K/V 32x128 bf16 tiles double-buffered in LDS,
// filled by the Tensor Data Mover (wave 0 issues, TENSORcnt-tracked) so tile
// i+1 streams in while tile i is consumed. Online softmax via per-wave LDS.
// ---------------------------------------------------------------------------
__global__ void __launch_bounds__(128)
flash_attn(const __bf16* __restrict__ Qb, const __bf16* __restrict__ Kb,
           const __bf16* __restrict__ Vb, __bf16* __restrict__ Ctx) {
  const int h     = blockIdx.x;
  const int kvh   = h >> 2;                // 4 q-heads per kv-head
  const int wave  = threadIdx.x >> 5;
  const int lane  = threadIdx.x & 31;
  const int qbase = blockIdx.y * 64 + wave * 16;

  __shared__ __align__(16) __bf16 Klds[2][32 * HDIM];     // 16 KB
  __shared__ __align__(16) __bf16 Vlds[2][32 * HDIM];     // 16 KB
  __shared__ __align__(16) float  Slds[4][16 * 32];       //  8 KB
  __shared__ __align__(16) __bf16 Plds[4][16 * 32];       //  4 KB
  __shared__ float Alds[4][16];
  __shared__ float Llds[4][16];

  // Q fragments for this wave's 16 rows (16x128 = 4 fragments of 16x32)
  v16bf qf[4];
#pragma unroll
  for (int i = 0; i < 4; ++i)
    qf[i] = load_frag_a(Qb + (size_t)h * HDIM, HID, qbase, i * 32);

  v8f acc[8];
#pragma unroll
  for (int j = 0; j < 8; ++j) acc[j] = (v8f){};
  float m_i = -3.0e38f, l_i = 0.0f;
  const float scale = 0.08838834764831845f;   // 1/sqrt(128)

  const int ntiles = SEQ / 32;

#if USE_TDM
  // preload tile 0 into buffer 0
  if (wave == 0) {
    tdm_load_tile_2d(&Klds[0][0], Kb + kvh * HDIM, 32, HDIM, KVDIM);
    tdm_load_tile_2d(&Vlds[0][0], Vb + kvh * HDIM, 32, HDIM, KVDIM);
  }
#endif

  for (int i = 0; i < ntiles; ++i) {
    const int p  = i & 1;
    const int kt = i * 32;
    __syncthreads();   // everyone done reading buffer 1-p (and, iter 0, Plds unused)

#if USE_TDM
    if (wave == 0) {
      if (i + 1 < ntiles) {
        const int ktn = (i + 1) * 32;
        tdm_load_tile_2d(&Klds[1 - p][0], Kb + (size_t)ktn * KVDIM + kvh * HDIM, 32, HDIM, KVDIM);
        tdm_load_tile_2d(&Vlds[1 - p][0], Vb + (size_t)ktn * KVDIM + kvh * HDIM, 32, HDIM, KVDIM);
        __builtin_amdgcn_s_wait_tensorcnt(2);   // tile i (2 older ops) complete
      } else {
        __builtin_amdgcn_s_wait_tensorcnt(0);
      }
    }
#else
    // fallback: cooperative vector copy of tile i into buffer p
#pragma unroll
    for (int t = 0; t < 4; ++t) {
      const int cid  = threadIdx.x + t * 128;  // 0..511
      const int rrow = cid >> 4;
      const int ccol = (cid & 15) * 8;
      *reinterpret_cast<v8bf*>(&Klds[p][rrow * HDIM + ccol]) =
        *reinterpret_cast<const v8bf*>(Kb + (size_t)(kt + rrow) * KVDIM + kvh * HDIM + ccol);
      *reinterpret_cast<v8bf*>(&Vlds[p][rrow * HDIM + ccol]) =
        *reinterpret_cast<const v8bf*>(Vb + (size_t)(kt + rrow) * KVDIM + kvh * HDIM + ccol);
    }
#endif
    __syncthreads();   // buffer p is filled and visible

    const __bf16* Kt = &Klds[p][0];
    const __bf16* Vt = &Vlds[p][0];

    // S = Q * K^T over D=128  (B[k=d][n=key] = Kt[key][d] -> NT loader)
    v8f s0 = {}, s1 = {};
#pragma unroll
    for (int ds = 0; ds < 4; ++ds) {
      v16bf kb0 = load_frag_b_nt(Kt, HDIM, 0,  ds * 32);
      v16bf kb1 = load_frag_b_nt(Kt, HDIM, 16, ds * 32);
      s0 = __builtin_amdgcn_wmma_f32_16x16x32_bf16(false, qf[ds], false, kb0, (short)0, s0, false, false);
      s1 = __builtin_amdgcn_wmma_f32_16x16x32_bf16(false, qf[ds], false, kb1, (short)0, s1, false, false);
    }

    // spill scaled scores to per-wave LDS buffer
    {
      const int rb = (lane < 16) ? 0 : 8;
      const int cc = lane & 15;
#pragma unroll
      for (int r = 0; r < 8; ++r) {
        Slds[wave][(rb + r) * 32 + cc]      = s0[r] * scale;
        Slds[wave][(rb + r) * 32 + 16 + cc] = s1[r] * scale;
      }
    }
    __syncthreads();

    // online softmax: lanes 0-15 each own one query row
    if (lane < 16) {
      const float* srow = &Slds[wave][lane * 32];
      float mnew = m_i;
#pragma unroll
      for (int j = 0; j < 32; ++j) mnew = fmaxf(mnew, srow[j]);
      const float alpha = __expf(m_i - mnew);
      float psum = 0.0f;
      __bf16* prow = &Plds[wave][lane * 32];
#pragma unroll
      for (int j = 0; j < 32; ++j) {
        const float pexp = __expf(srow[j] - mnew);
        psum += pexp;
        prow[j] = (__bf16)pexp;
      }
      l_i = l_i * alpha + psum;
      m_i = mnew;
      Alds[wave][lane] = alpha;
    }
    __syncthreads();

    // rescale running context accumulators by alpha(row)
    {
      const int rb = (lane < 16) ? 0 : 8;
      float a[8];
#pragma unroll
      for (int r = 0; r < 8; ++r) a[r] = Alds[wave][rb + r];
#pragma unroll
      for (int j = 0; j < 8; ++j)
#pragma unroll
        for (int r = 0; r < 8; ++r) acc[j][r] *= a[r];
    }

    // ctx += P * V   (A = P 16x32, B = V tile 32x128 column-gathered)
    v16bf pf = load_frag_a(&Plds[wave][0], 32, 0, 0);
#pragma unroll
    for (int j = 0; j < 8; ++j) {
      v16bf vf = load_frag_b_colmajor(Vt, HDIM, j * 16, 0);
      acc[j] = __builtin_amdgcn_wmma_f32_16x16x32_bf16(false, pf, false, vf, (short)0, acc[j], false, false);
    }
  }

  // epilogue: normalize by row-sum and store bf16 context [S, NH*D]
  __syncthreads();
  if (lane < 16) Llds[wave][lane] = l_i;
  __syncthreads();
  {
    const int rb = (lane < 16) ? 0 : 8;
    const int cc = lane & 15;
#pragma unroll
    for (int r = 0; r < 8; ++r) {
      const float inv = 1.0f / Llds[wave][rb + r];
      const int row = qbase + rb + r;
#pragma unroll
      for (int j = 0; j < 8; ++j)
        Ctx[(size_t)row * HID + h * HDIM + j * 16 + cc] = (__bf16)(acc[j][r] * inv);
    }
  }
}

// ---------------------------------------------------------------------------
// Host-side launch
// ---------------------------------------------------------------------------
extern "C" void kernel_launch(void* const* d_in, const int* in_sizes, int n_in,
                              void* d_out, int out_size, void* d_ws, size_t ws_size,
                              hipStream_t stream) {
  (void)in_sizes; (void)n_in; (void)out_size; (void)ws_size;

  const float* x    = (const float*)d_in[0];   // [1,2048,4096]
  const float* cosp = (const float*)d_in[1];   // [1,2048,128]
  const float* sinp = (const float*)d_in[2];   // [1,2048,128]
  const float* wq   = (const float*)d_in[3];   // [4096,4096]
  const float* wk   = (const float*)d_in[4];   // [1024,4096]
  const float* wv   = (const float*)d_in[5];   // [1024,4096]
  const float* wo   = (const float*)d_in[6];   // [4096,4096]
  float* out = (float*)d_out;                  // [1,2048,4096]

  char* ws = (char*)d_ws;
  const size_t MB = 1024 * 1024;
  __bf16* Xb   = (__bf16*)(ws);                // 16 MB
  __bf16* Wqb  = (__bf16*)(ws + 16  * MB);     // 32 MB
  __bf16* Wkb  = (__bf16*)(ws + 48  * MB);     //  8 MB
  __bf16* Wvb  = (__bf16*)(ws + 56  * MB);     //  8 MB
  __bf16* Wob  = (__bf16*)(ws + 64  * MB);     // 32 MB
  float*  Qf   = (float*) (ws + 96  * MB);     // 32 MB
  float*  Kf   = (float*) (ws + 128 * MB);     //  8 MB
  __bf16* Qb   = (__bf16*)(ws + 136 * MB);     // 16 MB
  __bf16* Kb   = (__bf16*)(ws + 152 * MB);     //  4 MB
  __bf16* Vb   = (__bf16*)(ws + 156 * MB);     //  4 MB
  __bf16* Ctxb = (__bf16*)(ws + 160 * MB);     // 16 MB  (total 176 MB)

  // 1) fp32 -> bf16 conversions
  auto cvt = [&](const float* src, __bf16* dst, size_t n) {
    int blocks = (int)((n + 1023) / 1024);
    cvt_f32_bf16<<<blocks, 256, 0, stream>>>(src, dst, n);
  };
  cvt(x,  Xb,  (size_t)SEQ * HID);
  cvt(wq, Wqb, (size_t)HID * HID);
  cvt(wk, Wkb, (size_t)KVDIM * HID);
  cvt(wv, Wvb, (size_t)KVDIM * HID);
  cvt(wo, Wob, (size_t)HID * HID);

  // 2) projections (NT GEMM; V straight to bf16, Q/K to fp32 for rope)
  gemm_nt_wmma<<<dim3(HID / 128,   SEQ / 64), 128, 0, stream>>>(Xb, Wqb, Qf, nullptr, SEQ, HID,   HID);
  gemm_nt_wmma<<<dim3(KVDIM / 128, SEQ / 64), 128, 0, stream>>>(Xb, Wkb, Kf, nullptr, SEQ, KVDIM, HID);
  gemm_nt_wmma<<<dim3(KVDIM / 128, SEQ / 64), 128, 0, stream>>>(Xb, Wvb, nullptr, Vb, SEQ, KVDIM, HID);

  // 3) rotary + bf16 repack
  {
    const int total = SEQ * (NHEADS + NKV) * (HDIM / 2);
    rope_pack<<<(total + 255) / 256, 256, 0, stream>>>(Qf, Kf, cosp, sinp, Qb, Kb);
  }

  // 4) flash attention (TDM double-buffered K/V tiles)
  flash_attn<<<dim3(NHEADS, SEQ / 64), 128, 0, stream>>>(Qb, Kb, Vb, Ctxb);

  // 5) output projection -> fp32 out
  gemm_nt_wmma<<<dim3(HID / 128, SEQ / 64), 128, 0, stream>>>(Ctxb, Wob, out, nullptr, SEQ, HID, HID);
}
